// QuantumLayer_34041910788367
// MI455X (gfx1250) — matrix operand order repaired and checked by reading, not compiled, not run on hardware
//
#include <hip/hip_runtime.h>
#include <hip/hip_bf16.h>

typedef __attribute__((ext_vector_type(16))) _Float16 v16h;
typedef __attribute__((ext_vector_type(8)))  float    v8f;

#define NQ    8
#define DIM   256      // 2^NQ
#define NLAY  2
#define MTILE 32       // samples per block (2 WMMA M-tiles)

// B-fragment storage: [kt(8)][nt(16)][lane(32)][h(16)] halves  (128 KB)
__device__ __forceinline__ int bfrag_idx(int kt, int nt, int lane, int h) {
    return ((kt * 16 + nt) * 32 + lane) * 16 + h;
}

// A-fragment LDS storage: [mt(2)][kt(8)][lane(32)][h(16)] halves (16 KB)
__device__ __forceinline__ int afrag_idx(int mt, int kt, int lane, int h) {
    return (((mt * 8 + kt) * 32) + lane) * 16 + h;
}

// -----------------------------------------------------------------------------
// Kernel 1: build the variational-layer operator U by applying the layer gates
// to each basis state. One column per wave: amplitude k = jj*32 + lane lives in
// register amp[jj]. Strides >= 32 are register permutes; strides < 32 use
// wave32 shfl_xor. Result stored as f16 pre-swizzled into WMMA B-fragment
// order (B[k][n] = U[n][k]).
// -----------------------------------------------------------------------------
__global__ void build_U_kernel(const float* __restrict__ weights,
                               _Float16* __restrict__ Bfrag) {
    const int lane = threadIdx.x & 31;
    const int wave = threadIdx.x >> 5;
    const int col  = blockIdx.x * (blockDim.x >> 5) + wave;   // basis index k

    float amp[8];
#pragma unroll
    for (int jj = 0; jj < 8; ++jj)
        amp[jj] = (jj * 32 + lane == col) ? 1.0f : 0.0f;

#pragma unroll
    for (int l = 0; l < NLAY; ++l) {
        // ---- RY(weights[l][i]) on every wire ----
#pragma unroll
        for (int i = 0; i < NQ; ++i) {
            const float th = 0.5f * weights[l * NQ + i];
            const float ct = cosf(th);
            const float st = sinf(th);
            const int   s  = 128 >> i;          // stride of wire i (bit 7-i)
            if (s >= 32) {                      // partner in another register
                const int r = s >> 5;
                float na[8];
#pragma unroll
                for (int jj = 0; jj < 8; ++jj) {
                    float v  = amp[jj];
                    float vp = amp[jj ^ r];
                    na[jj] = (jj & r) ? (st * vp + ct * v)
                                      : (ct * v - st * vp);
                }
#pragma unroll
                for (int jj = 0; jj < 8; ++jj) amp[jj] = na[jj];
            } else {                            // partner in another lane
#pragma unroll
                for (int jj = 0; jj < 8; ++jj) {
                    float v  = amp[jj];
                    float vp = __shfl_xor(v, s, 32);
                    amp[jj] = (lane & s) ? (st * vp + ct * v)
                                         : (ct * v - st * vp);
                }
            }
        }
        // ---- CNOT chain (control i, target i+1) ----
#pragma unroll
        for (int i = 0; i < NQ - 1; ++i) {
            const int sc  = 128 >> i;           // control stride
            const int stt = sc >> 1;            // target stride
            float na[8];
            if (stt >= 32) {
                const int r = stt >> 5;
#pragma unroll
                for (int jj = 0; jj < 8; ++jj) na[jj] = amp[jj ^ r];
            } else {
#pragma unroll
                for (int jj = 0; jj < 8; ++jj) na[jj] = __shfl_xor(amp[jj], stt, 32);
            }
            if (sc >= 32) {
                const int rc = sc >> 5;
#pragma unroll
                for (int jj = 0; jj < 8; ++jj)
                    if (jj & rc) amp[jj] = na[jj];
            } else {
#pragma unroll
                for (int jj = 0; jj < 8; ++jj)
                    amp[jj] = (lane & sc) ? na[jj] : amp[jj];
            }
        }
    }

    // ---- store column col into B-fragment layout ----
    // B layout used here (contiguous k, same as A): element (K,n) at lane = (n&15)+16*((K%32)>>4),
    // h = K&15, tile kt = K>>5, nt = n>>4.
    const int kt = col >> 5;
    const int kk = col & 31;
    const int g  = (kk >> 4) & 1;
    const int h  = kk & 15;
#pragma unroll
    for (int jj = 0; jj < 8; ++jj) {
        const int j  = jj * 32 + lane;
        const int nt = j >> 4;
        const int lp = (j & 15) + (g << 4);
        Bfrag[bfrag_idx(kt, nt, lp, h)] = (_Float16)amp[jj];
    }
}

// -----------------------------------------------------------------------------
// Kernel 2: per block, encode 32 samples into A fragments (Kron product of
// 8 cos/sin pairs), run the 32x256 @ 256x256 GEMM with v_wmma_f32_16x16x32_f16
// (8 waves x 2 N-tiles x 8 K-steps x 2 M-tiles; each B fragment held in
// registers feeds both M-tiles), square, and do the signed <Z_i> reduction.
// -----------------------------------------------------------------------------
__global__ void __launch_bounds__(256)
qcircuit_kernel(const float* __restrict__ x,
                const _Float16* __restrict__ Bfrag,
                float* __restrict__ out) {
    __shared__ __attribute__((aligned(32))) _Float16 sA[2 * 8 * 32 * 16]; // 16 KB
    __shared__ float pbuf[MTILE * DIM];                                   // 32 KB

    const int t  = threadIdx.x;
    const int b0 = blockIdx.x * MTILE;

    // ---- encode: thread t -> sample m = t&31, amplitudes k in [seg*32, seg*32+32)
    {
        const int m      = t & 31;
        const int seg    = t >> 5;          // 0..7, covers k bits 7..5 (wires 0..2)
        const int sample = b0 + m;
        const int mt     = m >> 4;
        const int mrow   = m & 15;
        float fc[8], fs[8];
#pragma unroll
        for (int i = 0; i < 8; ++i) {
            const float hx = 0.5f * x[sample * 8 + i];
            fc[i] = cosf(hx);
            fs[i] = sinf(hx);
        }
        float ph = 1.0f;
#pragma unroll
        for (int i = 0; i < 3; ++i)
            ph *= ((seg >> (2 - i)) & 1) ? fs[i] : fc[i];
#pragma unroll
        for (int low = 0; low < 32; ++low) {
            float p = ph;
#pragma unroll
            for (int i = 3; i < 8; ++i)
                p *= ((low >> (7 - i)) & 1) ? fs[i] : fc[i];
            const int k  = seg * 32 + low;
            const int kt = k >> 5;
            const int kk = k & 31;
            // A layout used here (contiguous k, same as B): K(h,L) = (L/16)*16 + h
            const int g    = (kk >> 4) & 1;
            const int lane = mrow + (g << 4);
            const int h    = kk & 15;
            sA[afrag_idx(mt, kt, lane, h)] = (_Float16)p;
        }
    }
    __syncthreads();

    // ---- GEMM: wave w covers output tiles nt = 2w, 2w+1 for both M-tiles
    const int w  = t >> 5;
    const int L  = t & 31;
    const int rb = (L >> 4) * 8;   // C-fragment row base: M = rb + r
    const int nl = L & 15;         // C-fragment column:  N = nt*16 + nl
#pragma unroll
    for (int nt2 = 0; nt2 < 2; ++nt2) {
        const int nt = w * 2 + nt2;
        v8f acc[2] = {{}, {}};
#pragma unroll
        for (int kt = 0; kt < 8; ++kt) {
            const v16h bm = *(const v16h*)&Bfrag[bfrag_idx(kt, nt, L, 0)];
#pragma unroll
            for (int mt = 0; mt < 2; ++mt) {
                const v16h a = *(const v16h*)&sA[afrag_idx(mt, kt, L, 0)];
                acc[mt] = __builtin_amdgcn_wmma_f32_16x16x32_f16(
                    /*neg_a=*/false, a, /*neg_b=*/false, bm,
                    /*c_mod=*/(short)0, acc[mt],
                    /*reuse_a=*/false, /*reuse_b=*/false);
            }
        }
#pragma unroll
        for (int mt = 0; mt < 2; ++mt) {
#pragma unroll
            for (int r = 0; r < 8; ++r) {
                const float v = acc[mt][r];
                pbuf[(mt * 16 + rb + r) * DIM + nt * 16 + nl] = v * v;
            }
        }
    }
    __syncthreads();

    // ---- <Z_i> = sum_j p[m][j] * (1 - 2*bit_{7-i}(j)), all 256 threads active
    {
        const int m     = t >> 3;      // 0..31
        const int i     = t & 7;
        const int shift = 7 - i;
        float z = 0.0f;
        for (int j = 0; j < DIM; ++j) {
            const float v = pbuf[m * DIM + j];
            z += ((j >> shift) & 1) ? -v : v;
        }
        out[(b0 + m) * NQ + i] = z;
    }
}

extern "C" void kernel_launch(void* const* d_in, const int* in_sizes, int n_in,
                              void* d_out, int out_size, void* d_ws, size_t ws_size,
                              hipStream_t stream) {
    const float* x       = (const float*)d_in[0];   // (131072, 8) f32
    const float* weights = (const float*)d_in[1];   // (2, 8) f32
    float*       out     = (float*)d_out;           // (131072, 8) f32
    _Float16*    Bfrag   = (_Float16*)d_ws;         // 128 KB swizzled U (f16)

    const int batch = in_sizes[0] / NQ;

    // 256 columns, one per wave: 32 blocks x 8 waves
    build_U_kernel<<<32, 256, 0, stream>>>(weights, Bfrag);

    // 32 samples per block
    qcircuit_kernel<<<batch / MTILE, 256, 0, stream>>>(x, Bfrag, out);
}
